// VectorQuantizer_41025527612071
// MI455X (gfx1250) — matrix operand order repaired and checked by reading, not compiled, not run on hardware
//
#include <hip/hip_runtime.h>

#define Dn   256
#define Kn   1024
#define HWn  1024
#define NTOT 32768
#define OUT_LOSS 8388608
#define OUT_PERP 8388609
#define OUT_IDX  8388610
#define XPAD 260   // fp32 LDS row stride: 260 mod 64 banks -> 4-bank step, conflict free

typedef __attribute__((ext_vector_type(16))) __bf16 v16bf;
typedef __attribute__((ext_vector_type(8)))  float  v8f;

__device__ __forceinline__ unsigned f2bf(float f) {
    unsigned u = __float_as_uint(f);
    u += 0x7FFFu + ((u >> 16) & 1u);   // round-to-nearest-even
    return u >> 16;
}

union Frag { v16bf v; unsigned u[8]; };

// --- Kernel 1: codebook -> bf16, -0.5*||e||^2, zero accumulators ---------------
__global__ void __launch_bounds__(256) vq_prep(const float* __restrict__ emb,
                                               unsigned short* __restrict__ ebf,
                                               float* __restrict__ enorm_h,
                                               float* __restrict__ counts,
                                               float* __restrict__ sse) {
    __shared__ float red[256];
    const int c = blockIdx.x, d = threadIdx.x;
    float v = emb[c * Dn + d];
    ebf[c * Dn + d] = (unsigned short)f2bf(v);
    red[d] = v * v;
    __syncthreads();
    for (int s = 128; s > 0; s >>= 1) {
        if (d < s) red[d] += red[d + s];
        __syncthreads();
    }
    if (d == 0) {
        enorm_h[c] = -0.5f * red[0];    // acc init: score = x.e - 0.5||e||^2
        counts[c]  = 0.0f;
        if (c == 0) sse[0] = 0.0f;
    }
}

// --- Kernel 2: fused score-GEMM (bf16 WMMA) + argmax + gather + SSE ------------
// 32 rows per block (two 16-row WMMA tiles share every B-fragment -> halves the
// L2 codebook traffic vs 16-row blocks). Argmin(dist) == argmax(score) since the
// accumulator is seeded with -0.5*||e||^2 (||x||^2 is row-constant, dropped).
__global__ void __launch_bounds__(256) vq_main(const float* __restrict__ x,
                                               const float* __restrict__ emb,
                                               const unsigned* __restrict__ ebf_u,
                                               const float* __restrict__ enorm_h,
                                               float* __restrict__ counts,
                                               float* __restrict__ sse,
                                               float* __restrict__ out) {
    __shared__ float xf[32 * XPAD];
    __shared__ float wbv[8][32];
    __shared__ int   wbi[8][32];
    __shared__ int   fidx[32];
    __shared__ float red[256];

    const int t   = threadIdx.x;
    const int n0  = blockIdx.x * 32;        // 32-row tile of flat [N,D]
    const int b   = n0 >> 10;               // 32 | HW, so one image per block
    const int hw0 = n0 & 1023;
    const float* xbase = x + (size_t)b * Dn * HWn + hw0;  // (m,d) at [d*HW + m]

    // Stage 32x256 fp32 X tile (128B coalesced runs along HW)
    {
        const int m = t & 31, dg = t >> 5;
        #pragma unroll
        for (int j = 0; j < 32; ++j) {
            const int d = dg + (j << 3);
            xf[m * XPAD + d] = xbase[d * HWn + m];
        }
    }
    __syncthreads();

    const int lane = t & 31;
    const int wave = t >> 5;
    const int m16  = lane & 15;
    const int kh   = lane >> 4;

    // Build A fragments for both row-tiles, all 8 k-steps (ISA 16-bit A layout)
    Frag a[2][8];
    #pragma unroll
    for (int rt = 0; rt < 2; ++rt) {
        const int mrow = rt * 16 + m16;
        #pragma unroll
        for (int ks = 0; ks < 8; ++ks) {
            #pragma unroll
            for (int j = 0; j < 8; ++j) {
                const int k = ks * 32 + ((j < 4) ? 0 : 16) + kh * 8 + (j & 3) * 2;
                const unsigned lo = f2bf(xf[mrow * XPAD + k]);
                const unsigned hi = f2bf(xf[mrow * XPAD + k + 1]);
                a[rt][ks].u[j] = lo | (hi << 16);
            }
        }
    }

    float bv0[8], bv1[8];
    int   bi0[8], bi1[8];
    #pragma unroll
    for (int r = 0; r < 8; ++r) {
        bv0[r] = -3.4e38f; bv1[r] = -3.4e38f; bi0[r] = 0; bi1[r] = 0;
    }

    // Each wave scans 8 code tiles of 16 codewords (codebook is L2-resident)
    for (int i = 0; i < 8; ++i) {
        const int c0  = (wave * 8 + i) * 16;
        const int col = c0 + m16;               // this lane's codeword column
        const float seed = enorm_h[col];        // -0.5*||e_col||^2
        v8f acc0, acc1;
        #pragma unroll
        for (int r = 0; r < 8; ++r) { acc0[r] = seed; acc1[r] = seed; }

        #pragma unroll
        for (int ks = 0; ks < 8; ++ks) {
            Frag bf;
            #pragma unroll
            for (int j = 0; j < 8; ++j) {
                const int k = ks * 32 + ((j < 4) ? 0 : 16) + kh * 8 + (j & 3) * 2;   // ISA 16-bit B layout
                bf.u[j] = ebf_u[(col * Dn + k) >> 1];
            }
            acc0 = __builtin_amdgcn_wmma_f32_16x16x32_bf16(
                false, a[0][ks].v, false, bf.v, (short)0, acc0, false, false);
            acc1 = __builtin_amdgcn_wmma_f32_16x16x32_bf16(
                false, a[1][ks].v, false, bf.v, (short)0, acc1, false, false);
        }
        // strict '>' keeps the lowest column on ties (ascending scan per lane)
        #pragma unroll
        for (int r = 0; r < 8; ++r) {
            if (acc0[r] > bv0[r]) { bv0[r] = acc0[r]; bi0[r] = col; }
            if (acc1[r] > bv1[r]) { bv1[r] = acc1[r]; bi1[r] = col; }
        }
    }

    // Max-reduce across the 16 lanes of each half (columns of the tile)
    #pragma unroll
    for (int off = 1; off < 16; off <<= 1) {
        #pragma unroll
        for (int r = 0; r < 8; ++r) {
            float ov = __shfl_xor(bv0[r], off);
            int   oi = __shfl_xor(bi0[r], off);
            if (ov > bv0[r] || (ov == bv0[r] && oi < bi0[r])) { bv0[r] = ov; bi0[r] = oi; }
            ov = __shfl_xor(bv1[r], off);
            oi = __shfl_xor(bi1[r], off);
            if (ov > bv1[r] || (ov == bv1[r] && oi < bi1[r])) { bv1[r] = ov; bi1[r] = oi; }
        }
    }
    if (m16 == 0) {
        #pragma unroll
        for (int r = 0; r < 8; ++r) {
            const int row = kh * 8 + r;
            wbv[wave][row]      = bv0[r];
            wbi[wave][row]      = bi0[r];
            wbv[wave][16 + row] = bv1[r];
            wbi[wave][16 + row] = bi1[r];
        }
    }
    __syncthreads();

    // Cross-wave reduce -> final argmax per row; emit indices + counts
    if (t < 32) {
        float bv = wbv[0][t]; int bi = wbi[0][t];
        #pragma unroll
        for (int w = 1; w < 8; ++w) {
            const float ov = wbv[w][t]; const int oi = wbi[w][t];
            if (ov > bv || (ov == bv && oi < bi)) { bv = ov; bi = oi; }
        }
        fidx[t] = bi;
        out[OUT_IDX + n0 + t] = (float)bi;
        atomicAdd(&counts[bi], 1.0f);
    }
    __syncthreads();

    // Gather exact fp32 codewords -> quantized_st output + SSE partial
    float accs = 0.0f;
    {
        const int mm = t & 31, dg = t >> 5;
        const int idx = fidx[mm];
        const float* e = emb + (size_t)idx * Dn;
        float* obase = out + (size_t)b * Dn * HWn + hw0;
        #pragma unroll
        for (int j = 0; j < 32; ++j) {
            const int d = dg + (j << 3);
            const float q  = e[d];
            const float xv = xf[mm * XPAD + d];
            obase[d * HWn + mm] = q;            // 128B coalesced along HW
            const float df = q - xv;
            accs += df * df;
        }
    }
    red[t] = accs;
    __syncthreads();
    for (int s = 128; s > 0; s >>= 1) {
        if (t < s) red[t] += red[t + s];
        __syncthreads();
    }
    if (t == 0) atomicAdd(sse, red[0]);
}

// --- Kernel 3: loss + perplexity ----------------------------------------------
__global__ void __launch_bounds__(256) vq_fin(const float* __restrict__ counts,
                                              const float* __restrict__ sse,
                                              float* __restrict__ out) {
    __shared__ float red[256];
    const int t = threadIdx.x;
    float s = 0.0f;
    for (int i = t; i < Kn; i += 256) {
        const float p = counts[i] * (1.0f / 32768.0f);
        s += p * logf(p + 1e-10f);
    }
    red[t] = s;
    __syncthreads();
    for (int st = 128; st > 0; st >>= 1) {
        if (t < st) red[t] += red[t + st];
        __syncthreads();
    }
    if (t == 0) {
        out[OUT_PERP] = expf(-red[0]);
        // q_latent + 0.25*e_latent both equal mean((q-x)^2) numerically
        out[OUT_LOSS] = 1.25f * sse[0] / 8388608.0f;
    }
}

extern "C" void kernel_launch(void* const* d_in, const int* in_sizes, int n_in,
                              void* d_out, int out_size, void* d_ws, size_t ws_size,
                              hipStream_t stream) {
    const float* x   = (const float*)d_in[0];   // [32,256,32,32] fp32
    const float* emb = (const float*)d_in[1];   // [1024,256] fp32
    float* out = (float*)d_out;

    char* ws = (char*)d_ws;
    unsigned short* ebf = (unsigned short*)ws;              // 1024*256*2 = 524288 B
    float* enorm_h = (float*)(ws + 524288);                 // 4 KB
    float* counts  = (float*)(ws + 524288 + 4096);          // 4 KB
    float* sse     = (float*)(ws + 524288 + 8192);          // 4 B

    vq_prep<<<Kn, 256, 0, stream>>>(emb, ebf, enorm_h, counts, sse);
    vq_main<<<NTOT / 32, 256, 0, stream>>>(x, emb, (const unsigned*)ebf,
                                           enorm_h, counts, sse, out);
    vq_fin<<<1, 256, 0, stream>>>(counts, sse, out);
}
